// LabelPredictor2D_69801808495255
// MI455X (gfx1250) — compile-verified, matrix operand order checked by reading, not run on hardware
//
#include <hip/hip_runtime.h>
#include <hip/hip_bf16.h>
#include <math.h>

typedef __attribute__((ext_vector_type(16))) __bf16 v16bf;
typedef __attribute__((ext_vector_type(8)))  float  v8f;

#define LDS_STRIDE 520  // 512 + 8 bf16 pad: keeps 16B alignment, avoids bank conflicts

// Packed-weight geometry: fragment = 16 bf16 per lane (32B), 32 lanes per k-step
// (1 KB), 16 k-steps per 16-wide N tile (16 KB). W1: 32 N-tiles, W2: 4 N-tiles
// (zero-padded to 64 labels).
#define TILE_BYTES 16384u                 // bytes per N-tile of packed weights
#define KSTEP_BYTES 1024u                 // bytes per k-step within a tile
#define W1_FRAGS (32 * 16 * 32)           // 16384 fragments
#define W2_FRAGS (4 * 16 * 32)            // 2048 fragments
#define W1_PACK_BYTES ((size_t)W1_FRAGS * 32)   // 512 KB
#define W2_PACK_BYTES ((size_t)W2_FRAGS * 32)   // 64 KB
#define WS_NEEDED (W1_PACK_BYTES + W2_PACK_BYTES)

// Load 8 f32 at p0 and 8 f32 at p1, convert to a 16-element bf16 fragment.
__device__ __forceinline__ v16bf cvt_pair_to_bf16(const float* p0, const float* p1) {
  const float4 a0 = *(const float4*)(p0);
  const float4 a1 = *(const float4*)(p0 + 4);
  const float4 c0 = *(const float4*)(p1);
  const float4 c1 = *(const float4*)(p1 + 4);
  v16bf r;
  r[0]  = (__bf16)a0.x; r[1]  = (__bf16)a0.y; r[2]  = (__bf16)a0.z; r[3]  = (__bf16)a0.w;
  r[4]  = (__bf16)a1.x; r[5]  = (__bf16)a1.y; r[6]  = (__bf16)a1.z; r[7]  = (__bf16)a1.w;
  r[8]  = (__bf16)c0.x; r[9]  = (__bf16)c0.y; r[10] = (__bf16)c0.z; r[11] = (__bf16)c0.w;
  r[12] = (__bf16)c1.x; r[13] = (__bf16)c1.y; r[14] = (__bf16)c1.z; r[15] = (__bf16)c1.w;
  return r;
}

// Read a 16-bf16 fragment from LDS in the interleaved A layout: 8 at p, 8 at p+16.
__device__ __forceinline__ v16bf lds_frag16(const __bf16* p) {
  union { uint4 u[2]; v16bf v; } x;
  x.u[0] = *(const uint4*)(p);
  x.u[1] = *(const uint4*)(p + 16);
  return x.v;
}

// Read a contiguous 32-byte packed fragment: uniform base + 32-bit lane offset so
// the backend can use the GVS (saddr + voffset) global_load_b128 form.
__device__ __forceinline__ v16bf packed_frag16(const char* base, uint32_t laneOff) {
  union { uint4 u[2]; v16bf v; } x;
  x.u[0] = *(const uint4*)(base + laneOff);
  x.u[1] = *(const uint4*)(base + laneOff + 16);
  return x.v;
}

// ---- Prep: convert W1/W2 (f32, row-major [n][k]) into bf16 WMMA-B fragments. ----
// Fragment f = (ntile*16 + kstep)*32 + lane; lane = half*16 + lr;
//   n = ntile*16 + lr,  k = kstep*32 + half*8 + {0..7, 16..23}.
__global__ __launch_bounds__(256)
void pack_weights_kernel(const float* __restrict__ W1,
                         const float* __restrict__ W2,
                         __bf16* __restrict__ pW1,
                         __bf16* __restrict__ pW2) {
  const int f = blockIdx.x * 256 + threadIdx.x;   // 0 .. W1_FRAGS+W2_FRAGS-1
  const bool isW2 = (f >= W1_FRAGS);
  const int fi   = isW2 ? (f - W1_FRAGS) : f;
  const int lane = fi & 31;
  const int ks   = (fi >> 5) & 15;
  const int nt   = fi >> 9;
  const int lr   = lane & 15;
  const int kA   = ks * 32 + (lane >> 4) * 8;
  const int n    = nt * 16 + lr;

  v16bf r;
  if (!isW2) {
    const float* w = W1 + (size_t)n * 512 + kA;
    r = cvt_pair_to_bf16(w, w + 16);
    *(v16bf*)(pW1 + (size_t)fi * 16) = r;
  } else {
    if (n < 50) {
      const float* w = W2 + (size_t)n * 512 + kA;
      r = cvt_pair_to_bf16(w, w + 16);
    } else {
      #pragma unroll
      for (int i = 0; i < 16; ++i) r[i] = (__bf16)0.0f;
    }
    *(v16bf*)(pW2 + (size_t)fi * 16) = r;
  }
}

// ---- Fused main: gather -> H = tanh(G @ W1^T + b1) -> out = H @ W2^T + b2 ----
// Grid: 64 blocks x 256 threads (8 wave32). Each block = one 16-row M tile.
// PACKED=true reads bf16 fragments from d_ws; PACKED=false converts f32 inline.
template <bool PACKED>
__global__ __launch_bounds__(256)
void label_mlp_fused_kernel(const float* __restrict__ feat,
                            const int*   __restrict__ heads,
                            const float* __restrict__ W1,
                            const float* __restrict__ b1,
                            const float* __restrict__ W2,
                            const float* __restrict__ b2,
                            const __bf16* __restrict__ pW1,
                            const __bf16* __restrict__ pW2,
                            float*       __restrict__ out) {
  __shared__ __bf16 ldsG[16 * LDS_STRIDE];  // gathered activations (bf16)
  __shared__ __bf16 ldsH[16 * LDS_STRIDE];  // hidden layer (bf16)

  const int tid   = threadIdx.x;
  const int wave  = tid >> 5;
  const int lane  = tid & 31;
  const int half  = lane >> 4;   // 0: K {0..7,16..23}; 1: K {8..15,24..31}
  const int lr    = lane & 15;   // row (A) / col (B) within the 16x16 tile
  const int mBase = blockIdx.x * 16;
  const uint32_t laneOff = (uint32_t)lane * 32u;  // per-lane byte offset in a fragment block

  // Phase 0: gather 16 rows of g = feat[b, l+1, heads[b,l], :] into LDS as bf16.
  {
    const int r = tid >> 4;          // 0..15 : row of the tile
    const int c = (tid & 15) * 32;   // 32 elements per thread
    const int m = mBase + r;
    __bf16* dst = &ldsG[r * LDS_STRIDE + c];
    if (m < 1016) {
      const int b  = m / 127;
      const int l  = m - b * 127;
      const int hd = heads[m];
      const float* src = feat + (((size_t)(b * 128 + (l + 1)) * 128 + (size_t)hd) << 9) + c;
      #pragma unroll
      for (int i = 0; i < 8; ++i) {
        const float4 v = *(const float4*)(src + i * 4);
        dst[i * 4 + 0] = (__bf16)v.x;
        dst[i * 4 + 1] = (__bf16)v.y;
        dst[i * 4 + 2] = (__bf16)v.z;
        dst[i * 4 + 3] = (__bf16)v.w;
      }
    } else {
      #pragma unroll
      for (int i = 0; i < 32; ++i) dst[i] = (__bf16)0.0f;
    }
  }
  __syncthreads();

  // Phase 1: GEMM1 (K=512, N=512). Each wave owns 4 N-tiles.
  {
    v8f acc[4];
    #pragma unroll
    for (int t = 0; t < 4; ++t) acc[t] = (v8f){0.f, 0.f, 0.f, 0.f, 0.f, 0.f, 0.f, 0.f};

    const int nt0 = wave * 4;   // first N tile owned by this wave

    if (PACKED) {
      // Uniform per-tile base pointers; per k-step advance is a scalar +1KB.
      const char* wb0 = (const char*)pW1 + (uint32_t)(nt0 + 0) * TILE_BYTES;
      const char* wb1 = (const char*)pW1 + (uint32_t)(nt0 + 1) * TILE_BYTES;
      const char* wb2 = (const char*)pW1 + (uint32_t)(nt0 + 2) * TILE_BYTES;
      const char* wb3 = (const char*)pW1 + (uint32_t)(nt0 + 3) * TILE_BYTES;

      // 2-stage software pipeline: prefetch k+1 fragments before the WMMA burst.
      v16bf bc0 = packed_frag16(wb0, laneOff);
      v16bf bc1 = packed_frag16(wb1, laneOff);
      v16bf bc2 = packed_frag16(wb2, laneOff);
      v16bf bc3 = packed_frag16(wb3, laneOff);

      for (int ks = 0; ks < 16; ++ks) {
        const int kA = ks * 32 + half * 8;
        const v16bf a = lds_frag16(&ldsG[lr * LDS_STRIDE + kA]);
        if (ks + 1 < 16) {
          const uint32_t nk = (uint32_t)(ks + 1) * KSTEP_BYTES;
          const v16bf bn0 = packed_frag16(wb0 + nk, laneOff);
          const v16bf bn1 = packed_frag16(wb1 + nk, laneOff);
          const v16bf bn2 = packed_frag16(wb2 + nk, laneOff);
          const v16bf bn3 = packed_frag16(wb3 + nk, laneOff);
          acc[0] = __builtin_amdgcn_wmma_f32_16x16x32_bf16(false, a, false, bc0, (short)0, acc[0], false, false);
          acc[1] = __builtin_amdgcn_wmma_f32_16x16x32_bf16(false, a, false, bc1, (short)0, acc[1], false, false);
          acc[2] = __builtin_amdgcn_wmma_f32_16x16x32_bf16(false, a, false, bc2, (short)0, acc[2], false, false);
          acc[3] = __builtin_amdgcn_wmma_f32_16x16x32_bf16(false, a, false, bc3, (short)0, acc[3], false, false);
          bc0 = bn0; bc1 = bn1; bc2 = bn2; bc3 = bn3;
        } else {
          acc[0] = __builtin_amdgcn_wmma_f32_16x16x32_bf16(false, a, false, bc0, (short)0, acc[0], false, false);
          acc[1] = __builtin_amdgcn_wmma_f32_16x16x32_bf16(false, a, false, bc1, (short)0, acc[1], false, false);
          acc[2] = __builtin_amdgcn_wmma_f32_16x16x32_bf16(false, a, false, bc2, (short)0, acc[2], false, false);
          acc[3] = __builtin_amdgcn_wmma_f32_16x16x32_bf16(false, a, false, bc3, (short)0, acc[3], false, false);
        }
      }
    } else {
      for (int ks = 0; ks < 16; ++ks) {
        const int kA = ks * 32 + half * 8;
        const v16bf a = lds_frag16(&ldsG[lr * LDS_STRIDE + kA]);
        #pragma unroll
        for (int t = 0; t < 4; ++t) {
          const float* w = W1 + (size_t)((nt0 + t) * 16 + lr) * 512 + kA;
          const v16bf bfrag = cvt_pair_to_bf16(w, w + 16);
          acc[t] = __builtin_amdgcn_wmma_f32_16x16x32_bf16(
              false, a, false, bfrag, (short)0, acc[t], false, false);
        }
      }
    }

    // Epilogue: bias + tanh, store hidden activations to LDS as bf16.
    #pragma unroll
    for (int t = 0; t < 4; ++t) {
      const int n = (nt0 + t) * 16 + lr;
      const float bias = b1[n];
      #pragma unroll
      for (int v = 0; v < 8; ++v) {
        const int mrow = v + half * 8;  // C layout: lanes 0-15 -> M=v, lanes 16-31 -> M=v+8
        ldsH[mrow * LDS_STRIDE + n] = (__bf16)tanhf(acc[t][v] + bias);
      }
    }
  }
  __syncthreads();

  // Phase 2: GEMM2 (K=512, N=50 padded to 64). Waves 0..3 each own one N-tile.
  if (wave < 4) {
    v8f acc = (v8f){0.f, 0.f, 0.f, 0.f, 0.f, 0.f, 0.f, 0.f};
    const int n    = wave * 16 + lr;  // label index 0..63
    const bool nOK = (n < 50);
    const char* wb = (const char*)pW2 + (uint32_t)wave * TILE_BYTES;

    for (int ks = 0; ks < 16; ++ks) {
      const int kA = ks * 32 + half * 8;
      const v16bf a = lds_frag16(&ldsH[lr * LDS_STRIDE + kA]);
      v16bf bfrag;
      if (PACKED) {
        bfrag = packed_frag16(wb + (uint32_t)ks * KSTEP_BYTES, laneOff);
      } else {
        if (nOK) {
          const float* w = W2 + (size_t)n * 512 + kA;
          bfrag = cvt_pair_to_bf16(w, w + 16);
        } else {
          #pragma unroll
          for (int i = 0; i < 16; ++i) bfrag[i] = (__bf16)0.0f;
        }
      }
      acc = __builtin_amdgcn_wmma_f32_16x16x32_bf16(
          false, a, false, bfrag, (short)0, acc, false, false);
    }

    if (nOK) {
      const float bias = b2[n];
      #pragma unroll
      for (int v = 0; v < 8; ++v) {
        const int m = mBase + v + half * 8;
        if (m < 1016) out[(size_t)m * 50 + n] = acc[v] + bias;
      }
    }
  }
}

extern "C" void kernel_launch(void* const* d_in, const int* in_sizes, int n_in,
                              void* d_out, int out_size, void* d_ws, size_t ws_size,
                              hipStream_t stream) {
  (void)in_sizes; (void)n_in; (void)out_size;
  const float* feat  = (const float*)d_in[0];
  const int*   heads = (const int*)d_in[1];
  const float* W1    = (const float*)d_in[2];
  const float* b1    = (const float*)d_in[3];
  const float* W2    = (const float*)d_in[4];
  const float* b2    = (const float*)d_in[5];
  float* out = (float*)d_out;

  if (ws_size >= WS_NEEDED && d_ws != nullptr) {
    __bf16* pW1 = (__bf16*)d_ws;
    __bf16* pW2 = (__bf16*)((char*)d_ws + W1_PACK_BYTES);
    pack_weights_kernel<<<(W1_FRAGS + W2_FRAGS) / 256, 256, 0, stream>>>(W1, W2, pW1, pW2);
    label_mlp_fused_kernel<true><<<64, 256, 0, stream>>>(
        feat, heads, W1, b1, W2, b2, pW1, pW2, out);
  } else {
    // Fallback: convert weights inline (no workspace required).
    label_mlp_fused_kernel<false><<<64, 256, 0, stream>>>(
        feat, heads, W1, b1, W2, b2, nullptr, nullptr, out);
  }
}
